// MLPScorer_49993419325442
// MI455X (gfx1250) — matrix operand order repaired and checked by reading, not compiled, not run on hardware
//
#include <hip/hip_runtime.h>

typedef __attribute__((ext_vector_type(16))) _Float16 v16h;
typedef __attribute__((ext_vector_type(8)))  float    v8f;
typedef __attribute__((ext_vector_type(4)))  float    v4f;
typedef __attribute__((ext_vector_type(4)))  int      v4i;

#define N_TOK 512
#define D_IN  512
#define H_HID 1024

#define AS1 __attribute__((address_space(1)))
#define AS3 __attribute__((address_space(3)))

// ---------------------------------------------------------------------------
// CDNA5 async global->LDS copy (ASYNCcnt-tracked), with synchronous fallback.
// Builtin signature (per hipcc diagnostic): (global v4i*, shared v4i*, Ii, Ii)
// ---------------------------------------------------------------------------
#if __has_builtin(__builtin_amdgcn_global_load_async_to_lds_b128) && \
    __has_builtin(__builtin_amdgcn_s_wait_asynccnt)
#define ASYNC_COPY 1
#else
#define ASYNC_COPY 0
#endif

__device__ __forceinline__ void copy_b128(float* lds, const float* g) {
#if ASYNC_COPY
    __builtin_amdgcn_global_load_async_to_lds_b128((AS1 v4i*)g, (AS3 v4i*)lds, 0, 0);
#else
    *(v4f*)lds = *(const v4f*)g;
#endif
}

__device__ __forceinline__ void wait_async() {
#if ASYNC_COPY
    __builtin_amdgcn_s_wait_asynccnt(0);
#endif
}

// ---------------------------------------------------------------------------
// fast tanh: native gfx1250 v_tanh_f32 if the toolchain exposes it, else
// exp2/rcp formulation (t = e^{2x}; tanh = (t-1)/(t+1)), clamped for safety.
// ---------------------------------------------------------------------------
__device__ __forceinline__ float tanh_fast(float x) {
#if __has_builtin(__builtin_amdgcn_tanhf)
    return __builtin_amdgcn_tanhf(x);
#else
    float xc = fminf(fmaxf(x, -10.0f), 10.0f);
    float t  = __builtin_amdgcn_exp2f(xc * 2.885390081777927f); // 2*log2(e)
    return (t - 1.0f) * __builtin_amdgcn_rcpf(t + 1.0f);
#endif
}

// ---------------------------------------------------------------------------
// Kernel A: HI = X @ W1[0:512,:] + b1   (mat==0)
//           HJ = X @ W1[512:1024,:]     (mat==1)
// One wave per 16x16 output tile, K-loop of 32 with split-f16 WMMA
// (al*bh + ah*bl + ah*bh) accumulated in f32 -> ~f32 accuracy.
// ---------------------------------------------------------------------------
__global__ __launch_bounds__(128)
void mlp_proj_wmma(const float* __restrict__ X,
                   const float* __restrict__ W1,
                   const float* __restrict__ b1,
                   float* __restrict__ OUT)       // [2][512][1024]
{
    const int lane    = threadIdx.x & 31;
    const int wave    = threadIdx.x >> 5;
    const int colTile = blockIdx.x * 4 + wave;    // 64 tiles of 16 cols
    const int rowTile = blockIdx.y;               // 32 tiles of 16 rows
    const int mat     = blockIdx.z;               // 0 = HI, 1 = HJ

    const int row0  = rowTile * 16;
    const int col0  = colTile * 16;
    const int wrow0 = mat * D_IN;                 // row offset into W1

    // A-fragment addressing (ISA 7.12.2, 16-bit A 16x32):
    //   lanes 0-15:  M=lane,    halves 0-7 -> K=0..7,   halves 8-15 -> K=16..23
    //   lanes 16-31: M=lane-16, halves 0-7 -> K=8..15,  halves 8-15 -> K=24..31
    const int aM    = lane & 15;
    const int kBase = (lane >> 4) << 3;           // 0 or 8

    v8f acc = {};
    for (int k0 = 0; k0 < D_IN; k0 += 32) {
        const float* arow = X + (size_t)(row0 + aM) * D_IN + k0;
        v4f a0 = *(const v4f*)(arow + kBase);
        v4f a1 = *(const v4f*)(arow + kBase + 4);
        v4f a2 = *(const v4f*)(arow + 16 + kBase);
        v4f a3 = *(const v4f*)(arow + 16 + kBase + 4);

        // B-fragment: N = lane&15, halves use same K order as A (kBase.., 16+kBase..)
        const float* brow = W1 + (size_t)(wrow0 + k0) * H_HID + col0 + aM;
        v4f b0, b1v, b2v, b3v;
#pragma unroll
        for (int e = 0; e < 4; ++e) {
            b0[e]  = brow[(size_t)(kBase + e) * H_HID];
            b1v[e] = brow[(size_t)(kBase + 4 + e) * H_HID];
            b2v[e] = brow[(size_t)(16 + kBase + e) * H_HID];
            b3v[e] = brow[(size_t)(16 + kBase + 4 + e) * H_HID];
        }

        float afv[16], bfv[16];
        *(v4f*)(afv + 0)  = a0;  *(v4f*)(afv + 4)  = a1;
        *(v4f*)(afv + 8)  = a2;  *(v4f*)(afv + 12) = a3;
        *(v4f*)(bfv + 0)  = b0;  *(v4f*)(bfv + 4)  = b1v;
        *(v4f*)(bfv + 8)  = b2v; *(v4f*)(bfv + 12) = b3v;

        v16h ah, al, bh, bl;
#pragma unroll
        for (int e = 0; e < 16; ++e) {
            _Float16 ha = (_Float16)afv[e];
            ah[e] = ha;
            al[e] = (_Float16)(afv[e] - (float)ha);
            _Float16 hb = (_Float16)bfv[e];
            bh[e] = hb;
            bl[e] = (_Float16)(bfv[e] - (float)hb);
        }

        // split-precision accumulation: a*b ~= al*bh + ah*bl + ah*bh
        acc = __builtin_amdgcn_wmma_f32_16x16x32_f16(false, al, false, bh,
                                                     (short)0, acc, false, false);
        acc = __builtin_amdgcn_wmma_f32_16x16x32_f16(false, ah, false, bl,
                                                     (short)0, acc, false, false);
        acc = __builtin_amdgcn_wmma_f32_16x16x32_f16(false, ah, false, bh,
                                                     (short)0, acc, false, false);
    }

    // C/D layout: VGPR r -> M = r + (lane<16 ? 0 : 8), N = lane&15
    const int cN     = lane & 15;
    const int cMbase = (lane >> 4) << 3;
    const float bias = (mat == 0) ? b1[col0 + cN] : 0.0f;  // fold b1 into HI
    float* outBase = OUT + (size_t)mat * N_TOK * H_HID;
#pragma unroll
    for (int r = 0; r < 8; ++r) {
        outBase[(size_t)(row0 + cMbase + r) * H_HID + col0 + cN] = acc[r] + bias;
    }
}

// ---------------------------------------------------------------------------
// Kernel B: scores[i,j] = b2 + sum_h tanh(HI[i,h] + HJ[j,h]) * W2[h]
// 32x32 output tile per 256-thread workgroup; H streamed in 128-wide chunks
// through double-buffered LDS via async global->LDS b128 copies (ASYNCcnt),
// overlapping the copy of chunk c+1 with the tanh/FMA compute of chunk c.
// Inner loop vectorized by 4 (ds_load_b128): row pad +4 dwords keeps 16B
// alignment; per-lane hj read is bank (4j+h)%64 -> only 2-way conflict.
// Dominant cost: 268M v_tanh_f32 (TRANS) co-executing with the FMA stream.
// ---------------------------------------------------------------------------
#define TILE 32
#define CH   128
#define CPAD 132
#define NCH  (H_HID / CH)

__global__ __launch_bounds__(256)
void mlp_pair_score(const float* __restrict__ HI,
                    const float* __restrict__ HJ,
                    const float* __restrict__ W2,
                    const float* __restrict__ b2,
                    float* __restrict__ OUT)      // [512][512]
{
    __shared__ float hi_s[2][TILE][CPAD];
    __shared__ float hj_s[2][TILE][CPAD];
    __shared__ float w2_s[H_HID];

    const int i0 = blockIdx.y * TILE;
    const int j0 = blockIdx.x * TILE;
    const int t  = threadIdx.x;
    const int j  = t & 31;        // lane -> j
    const int r  = t >> 5;        // wave id -> i rows {r, r+8, r+16, r+24}

    // stage a 32x128 chunk of HI and HJ into LDS buffer p (b128 granules)
    auto stage = [&](int p, int h0) {
#pragma unroll
        for (int q = 0; q < (TILE * CH) / (4 * 256); ++q) {   // 4 iters
            int g   = t + 256 * q;
            int row = g >> 5;
            int c4  = (g & 31) << 2;
            copy_b128(&hi_s[p][row][c4],
                      HI + (size_t)(i0 + row) * H_HID + h0 + c4);
            copy_b128(&hj_s[p][row][c4],
                      HJ + (size_t)(j0 + row) * H_HID + h0 + c4);
        }
    };

    // W2 -> LDS once (synchronous, coalesced)
#pragma unroll
    for (int q = 0; q < H_HID / 256; ++q)
        w2_s[t + 256 * q] = W2[t + 256 * q];

    stage(0, 0);
    wait_async();
    __syncthreads();

    v4f acc0 = {}, acc1 = {}, acc2 = {}, acc3 = {};

    for (int c = 0; c < NCH; ++c) {
        const int cur = c & 1;
        if (c + 1 < NCH) stage(cur ^ 1, (c + 1) * CH);   // prefetch next chunk

#pragma unroll 4
        for (int h4 = 0; h4 < CH; h4 += 4) {
            v4f w   = *(const v4f*)&w2_s[c * CH + h4];       // uniform
            v4f hjv = *(const v4f*)&hj_s[cur][j][h4];        // per-lane
            v4f x0  = *(const v4f*)&hi_s[cur][r     ][h4];   // wave-uniform
            v4f x1  = *(const v4f*)&hi_s[cur][r +  8][h4];
            v4f x2  = *(const v4f*)&hi_s[cur][r + 16][h4];
            v4f x3  = *(const v4f*)&hi_s[cur][r + 24][h4];
            x0 += hjv; x1 += hjv; x2 += hjv; x3 += hjv;
#pragma unroll
            for (int e = 0; e < 4; ++e) {
                acc0[e] += tanh_fast(x0[e]) * w[e];
                acc1[e] += tanh_fast(x1[e]) * w[e];
                acc2[e] += tanh_fast(x2[e]) * w[e];
                acc3[e] += tanh_fast(x3[e]) * w[e];
            }
        }

        wait_async();      // prefetch of next chunk has landed in LDS
        __syncthreads();   // visible to all waves before buffer swap
    }

    const float bb = b2[0];
    const float s0 = acc0[0] + acc0[1] + acc0[2] + acc0[3];
    const float s1 = acc1[0] + acc1[1] + acc1[2] + acc1[3];
    const float s2 = acc2[0] + acc2[1] + acc2[2] + acc2[3];
    const float s3 = acc3[0] + acc3[1] + acc3[2] + acc3[3];
    OUT[(size_t)(i0 + r     ) * N_TOK + j0 + j] = s0 + bb;
    OUT[(size_t)(i0 + r +  8) * N_TOK + j0 + j] = s1 + bb;
    OUT[(size_t)(i0 + r + 16) * N_TOK + j0 + j] = s2 + bb;
    OUT[(size_t)(i0 + r + 24) * N_TOK + j0 + j] = s3 + bb;
}

// ---------------------------------------------------------------------------
extern "C" void kernel_launch(void* const* d_in, const int* in_sizes, int n_in,
                              void* d_out, int out_size, void* d_ws, size_t ws_size,
                              hipStream_t stream) {
    const float* X  = (const float*)d_in[0];   // [512, 512]
    const float* W1 = (const float*)d_in[1];   // [1024, 1024]
    const float* b1 = (const float*)d_in[2];   // [1024]
    const float* W2 = (const float*)d_in[3];   // [1024, 1]
    const float* b2 = (const float*)d_in[4];   // [1]
    float* OUT = (float*)d_out;                // [512, 512]

    float* HIJ = (float*)d_ws;                 // [2][512][1024] f32 = 4 MB
    const float* HI = HIJ;
    const float* HJ = HIJ + (size_t)N_TOK * H_HID;

    dim3 gA(H_HID / 16 / 4, N_TOK / 16, 2);    // (16, 32, 2), 128 thr = 4 waves
    mlp_proj_wmma<<<gA, 128, 0, stream>>>(X, W1, b1, HIJ);

    dim3 gB(N_TOK / TILE, N_TOK / TILE, 1);    // (16, 16)
    mlp_pair_score<<<gB, 256, 0, stream>>>(HI, HJ, W2, b2, OUT);
}
